// ContrastiveHead_42314017800610
// MI455X (gfx1250) — matrix operand order repaired and checked by reading, not compiled, not run on hardware
//
#include <hip/hip_runtime.h>
#include <hip/hip_bf16.h>
#include <hip/hip_fp16.h>

// CDNA5 / gfx1250, wave32.
typedef __attribute__((ext_vector_type(16))) _Float16 v16h;
typedef __attribute__((ext_vector_type(8)))  _Float16 v8h;
typedef __attribute__((ext_vector_type(8)))  float    v8f;

#define N_TOT 8192      // 2*B
#define HALF_N 4096     // B
#define DIM   256
#define INV_T 10.0f     // 1/temperature
#define SHIFT 10.0f     // fixed logsumexp shift (sim/T <= 10 since |sim|<=1)
#define PANEL_COLS 64   // columns staged in LDS per buffer (32 KB fp16)
#define PANEL_BYTES (PANEL_COLS * DIM * 2)   // 32768
#define NPANELS (N_TOT / PANEL_COLS)         // 128

// gfx1250 async global->LDS path (ASYNCcnt-tracked DMA, no VGPR staging).
// Probe-derived signature: (AS1 int4* src, AS3 int4* dst, Imm off, Imm cpol),
// pointee = GCC-style vector_size(16) int, non-const.
#if __has_builtin(__builtin_amdgcn_global_load_async_to_lds_b128)
#define USE_ASYNC_LDS 1
typedef int v4i_gcc __attribute__((vector_size(16)));
typedef __attribute__((address_space(1))) v4i_gcc as1_v4i;
typedef __attribute__((address_space(3))) v4i_gcc as3_v4i;
#else
#define USE_ASYNC_LDS 0
#endif

static __device__ __forceinline__ void wait_asynccnt0() {
#if __has_builtin(__builtin_amdgcn_s_wait_asynccnt)
  __builtin_amdgcn_s_wait_asynccnt(0);
#elif USE_ASYNC_LDS
  asm volatile("s_wait_asynccnt 0" ::: "memory");
#endif
}

// ---------------------------------------------------------------------------
// Kernel 1: L2-normalize each row of concat(q,k) and write fp16 rows to ws.
// One wave32 per row: lane holds 8 consecutive floats.
// ---------------------------------------------------------------------------
__global__ __launch_bounds__(256)
void nrm_kernel(const float* __restrict__ q, const float* __restrict__ k,
                _Float16* __restrict__ fh) {
  int row  = (blockIdx.x * blockDim.x + threadIdx.x) >> 5;
  int lane = threadIdx.x & 31;
  if (row >= N_TOT) return;
  const float* src = (row < HALF_N) ? (q + (size_t)row * DIM)
                                    : (k + (size_t)(row - HALF_N) * DIM);
  float4 x0 = ((const float4*)src)[lane * 2 + 0];
  float4 x1 = ((const float4*)src)[lane * 2 + 1];
  float ss = x0.x*x0.x + x0.y*x0.y + x0.z*x0.z + x0.w*x0.w
           + x1.x*x1.x + x1.y*x1.y + x1.z*x1.z + x1.w*x1.w;
#pragma unroll
  for (int m = 16; m >= 1; m >>= 1) ss += __shfl_xor(ss, m, 32);
  float nrm = fmaxf(sqrtf(ss), 1e-12f);
  float inv = 1.0f / nrm;
  v8h o;
  o[0] = (_Float16)(x0.x * inv); o[1] = (_Float16)(x0.y * inv);
  o[2] = (_Float16)(x0.z * inv); o[3] = (_Float16)(x0.w * inv);
  o[4] = (_Float16)(x1.x * inv); o[5] = (_Float16)(x1.y * inv);
  o[6] = (_Float16)(x1.z * inv); o[7] = (_Float16)(x1.w * inv);
  *(v8h*)(fh + (size_t)row * DIM + lane * 8) = o;
}

// ---------------------------------------------------------------------------
// Panel stage: 64 rows x 256 halfs (32 KB) into an LDS buffer.
// Async path: each thread issues 8 b128 DMA transfers (strided 4 KB).
// ---------------------------------------------------------------------------
static __device__ __forceinline__ void issue_panel_load(
    const _Float16* __restrict__ src, _Float16* __restrict__ dst, int tid) {
#if USE_ASYNC_LDS
  char* g = (char*)src + tid * 16;   // our own ws; builtin wants non-const
  char* l = (char*)dst + tid * 16;
#pragma unroll
  for (int j = 0; j < PANEL_BYTES / (256 * 16); ++j) {
    __builtin_amdgcn_global_load_async_to_lds_b128(
        (as1_v4i*)(g + j * 4096), (as3_v4i*)(l + j * 4096),
        /*offset=*/0, /*cpol=*/0);
  }
#else
  const uint4* s = (const uint4*)src;
  uint4*       d = (uint4*)dst;
  for (int i = tid; i < PANEL_BYTES / 16; i += 256) d[i] = s[i];
#endif
}

// ---------------------------------------------------------------------------
// Kernel 2: fused sim-GEMM + per-row sum(exp(sim/T - SHIFT)) + pos capture.
// Block = 8 waves; wave w owns rows [blk*128 + w*16, +16). Double-buffered
// 64-column panels of f staged in LDS (async DMA overlapped with compute),
// shared by all 8 waves. 16x16x32 f16 WMMA, fp32 accumulate.
// ---------------------------------------------------------------------------
__global__ __launch_bounds__(256)
void simclr_kernel(const _Float16* __restrict__ fh, float* __restrict__ rowloss) {
  __shared__ _Float16 panel[2][PANEL_COLS * DIM];   // 2 x 32 KB

  const int tid  = threadIdx.x;
  const int w    = tid >> 5;
  const int lane = tid & 31;
  const int hi   = lane >> 4;       // 0: lanes 0-15, 1: lanes 16-31
  const int l15  = lane & 15;
  const int mbase = blockIdx.x * 128 + w * 16;

  // --- Load A fragments (ISA 16-bit A 16x32 layout): row = mbase + l15.
  // element e in [0,8):  k = ktile*32 + hi*8 + e
  // element e in [8,16): k = ktile*32 + hi*8 + 16 + (e-8)
  v16h A[8];
  {
    const _Float16* row = fh + (size_t)(mbase + l15) * DIM;
#pragma unroll
    for (int kt = 0; kt < 8; ++kt) {
      v8h lo = *(const v8h*)(row + kt * 32 + hi * 8);
      v8h hh = *(const v8h*)(row + kt * 32 + hi * 8 + 16);
#pragma unroll
      for (int e = 0; e < 8; ++e) { A[kt][e] = lo[e]; A[kt][8 + e] = hh[e]; }
    }
  }

  float sumexp[8], pos[8];
#pragma unroll
  for (int r = 0; r < 8; ++r) { sumexp[r] = 0.0f; pos[r] = 0.0f; }

  // Prologue: stage panel 0 into buffer 0.
  issue_panel_load(fh, panel[0], tid);
  wait_asynccnt0();
  __syncthreads();

  for (int p = 0; p < NPANELS; ++p) {
    const int cur = p & 1;
    const int c0  = p * PANEL_COLS;

    // Prefetch next panel into the other buffer; its previous readers all
    // passed the barrier at the end of iteration p-1, so it is free.
    if (p + 1 < NPANELS)
      issue_panel_load(fh + (size_t)(c0 + PANEL_COLS) * DIM, panel[cur ^ 1], tid);

#pragma unroll
    for (int nt = 0; nt < 4; ++nt) {
      // B fragment (ISA B KxN layout): lane column n = l15,
      // element e in [0,8): k = ktile*32 + hi*8 + e; e in [8,16): k = ktile*32 + hi*8 + 16 + (e-8).
      const _Float16* bb = panel[cur] + (size_t)(nt * 16 + l15) * DIM + hi * 8;
      v8f acc = {};
#pragma unroll
      for (int kt = 0; kt < 8; ++kt) {
        v8h blo = *(const v8h*)(bb + kt * 32);
        v8h bhi = *(const v8h*)(bb + kt * 32 + 16);
        v16h B = __builtin_shufflevector(blo, bhi, 0, 1, 2, 3, 4, 5, 6, 7, 8, 9, 10, 11, 12, 13, 14, 15);
        acc = __builtin_amdgcn_wmma_f32_16x16x32_f16(
            /*neg_a=*/false, A[kt], /*neg_b=*/false, B,
            /*c_mod=*/(short)0, acc, /*reuse_a=*/false, /*reuse_b=*/false);
      }
      // C/D layout: c[r] -> m = mbase + r + hi*8, n = c0 + nt*16 + l15.
      const int n  = c0 + nt * 16 + l15;
      const int mb = mbase + hi * 8;
#pragma unroll
      for (int r = 0; r < 8; ++r) {
        const int m = mb + r;
        const float s = acc[r];
        if (n != m) sumexp[r] += __expf(s * INV_T - SHIFT);
        const int partner = (m + HALF_N) & (N_TOT - 1);
        if (n == partner) pos[r] = s;
      }
    }

    // My prefetch DMAs must land, then block-wide rendezvous: after this
    // barrier everyone's next-panel data is visible and the old buffer is
    // reusable.
    wait_asynccnt0();
    __syncthreads();
  }

  // Reduce across the 16 N-lanes of each half-wave (masks <= 8 keep the
  // two M-groups {lanes 0-15} / {lanes 16-31} separate).
#pragma unroll
  for (int r = 0; r < 8; ++r) {
#pragma unroll
    for (int msk = 8; msk >= 1; msk >>= 1) {
      sumexp[r] += __shfl_xor(sumexp[r], msk, 32);
      pos[r]    += __shfl_xor(pos[r],    msk, 32);
    }
  }
  if (l15 == 0) {
    const int mb = mbase + hi * 8;
#pragma unroll
    for (int r = 0; r < 8; ++r)
      rowloss[mb + r] = __logf(sumexp[r]) + SHIFT - pos[r] * INV_T;
  }
}

// ---------------------------------------------------------------------------
// Kernel 3: deterministic single-block mean over 8192 per-row losses.
// ---------------------------------------------------------------------------
__global__ __launch_bounds__(256)
void reduce_kernel(const float* __restrict__ rowloss, float* __restrict__ out) {
  __shared__ float sb[256];
  float s = 0.0f;
  for (int i = threadIdx.x; i < N_TOT; i += 256) s += rowloss[i];
  sb[threadIdx.x] = s;
  __syncthreads();
  for (int step = 128; step > 0; step >>= 1) {
    if ((int)threadIdx.x < step) sb[threadIdx.x] += sb[threadIdx.x + step];
    __syncthreads();
  }
  if (threadIdx.x == 0) out[0] = sb[0] * (1.0f / (float)N_TOT);
}

// ---------------------------------------------------------------------------
extern "C" void kernel_launch(void* const* d_in, const int* in_sizes, int n_in,
                              void* d_out, int out_size, void* d_ws, size_t ws_size,
                              hipStream_t stream) {
  const float* q = (const float*)d_in[0];
  const float* k = (const float*)d_in[1];
  _Float16* fh      = (_Float16*)d_ws;                                   // 4 MB
  float*    rowloss = (float*)((char*)d_ws +
                               (size_t)N_TOT * DIM * sizeof(_Float16));  // 32 KB
  float* out = (float*)d_out;

  nrm_kernel   <<<N_TOT / 8,   256, 0, stream>>>(q, k, fh);
  simclr_kernel<<<N_TOT / 128, 256, 0, stream>>>(fh, rowloss);
  reduce_kernel<<<1,           256, 0, stream>>>(rowloss, out);
}
